// BidirectionalTemporalAttention_33844342292640
// MI455X (gfx1250) — compile-verified, matrix operand order checked by reading, not run on hardware
//
#include <hip/hip_runtime.h>

// ---------------------------------------------------------------------------
// BidirectionalTemporalAttention on gfx1250 (MI455X, wave32, WMMA + TDM)
// N=4096 tokens, D=768, H=12 heads x 64, heads 0..5 causal, 6..11 anti-causal
// All GEMMs via v_wmma_f32_16x16x32_bf16; flash softmax; TDM weight staging.
// ---------------------------------------------------------------------------

typedef __bf16 bf16_t;
typedef bf16_t v4bf  __attribute__((ext_vector_type(4)));
typedef bf16_t v8bf  __attribute__((ext_vector_type(8)));
typedef bf16_t v16bf __attribute__((ext_vector_type(16)));
typedef float  v8f   __attribute__((ext_vector_type(8)));
typedef unsigned int u32x4 __attribute__((ext_vector_type(4)));
typedef int          i32x8 __attribute__((ext_vector_type(8)));
typedef int          i32x4 __attribute__((ext_vector_type(4)));

#define NSEQ   4096
#define DMODEL 768
#define NHEAD  12
#define HDIM   64
#define NLOOKB 6

#if defined(__has_builtin)
#if __has_builtin(__builtin_amdgcn_tensor_load_to_lds) && __has_builtin(__builtin_amdgcn_s_wait_tensorcnt)
#define USE_TDM 1
#endif
#endif
#ifndef USE_TDM
#define USE_TDM 0
#endif

// Padded-LDS weight tile: logical [64][768] bf16 stream, with 4 dwords of pad
// inserted every 128 dwords (TDM pad_interval=6, pad_amount=3).
// dword s  ->  lds byte  4*s + 16*(s>>7).   Effective row stride 396 dwords
// == 12 mod 64 banks -> 16 B-fragment lanes land on 16 disjoint 4-bank groups.
#define WTILE_BYTES ((HDIM * DMODEL * 2) + ((HDIM * DMODEL / 2 / 128) * 16))  // 101376

__device__ __forceinline__ const bf16_t* lds_pad_ptr(const bf16_t* base, int elem) {
    int s = elem >> 1;                               // elem is always even here
    int byteoff = (s << 2) + ((s >> 7) << 4);
    return (const bf16_t*)((const char*)base + byteoff);
}

// Load one 16-bit A/B fragment (16 bf16 = 8 VGPRs per lane).
// p = &row[kchunk + 8*(lane>>4)]; hi half at +16 elements (ISA 7.12.2 layout).
__device__ __forceinline__ v16bf load_frag(const bf16_t* p) {
    v8bf lo = *(const v8bf*)(p);
    v8bf hi = *(const v8bf*)(p + 16);
    v16bf r;
#pragma unroll
    for (int i = 0; i < 8; ++i) { r[i] = lo[i]; r[i + 8] = hi[i]; }
    return r;
}

__device__ __forceinline__ v16bf load_frag2(const bf16_t* plo, const bf16_t* phi) {
    v8bf lo = *(const v8bf*)(plo);
    v8bf hi = *(const v8bf*)(phi);
    v16bf r;
#pragma unroll
    for (int i = 0; i < 8; ++i) { r[i] = lo[i]; r[i + 8] = hi[i]; }
    return r;
}

__device__ __forceinline__ v8f wmma_bf16(v16bf a, v16bf b, v8f c) {
    return __builtin_amdgcn_wmma_f32_16x16x32_bf16(
        false, a, false, b, (short)0, c, false, false);
}

// Stage one contiguous [64][768] bf16 weight tile into padded LDS.
// TDM path: 1D copy, data_size=8B, n8=12288 elements, LDS padding in D#.
__device__ __forceinline__ void stage_wtile(bf16_t* wlds, const bf16_t* wt) {
#if USE_TDM
    if (threadIdx.x == 0) {
        unsigned long long ga = (unsigned long long)(uintptr_t)wt;
        unsigned lds_off = (unsigned)(uintptr_t)(void*)wlds;   // low 32 bits = LDS offset
        const unsigned n8 = (HDIM * DMODEL * 2) / 8;           // 12288 x 8B
        u32x4 g0;
        g0[0] = 1u;                                            // count=1 (valid, user)
        g0[1] = lds_off;                                       // lds_addr
        g0[2] = (unsigned)(ga & 0xFFFFFFFFu);                  // global_addr[31:0]
        g0[3] = (unsigned)((ga >> 32) & 0x01FFFFFFu) | (2u << 30); // addr[56:32]|type=2
        i32x8 g1;
        g1[0] = (3 << 16) | (1 << 20) | (6 << 22) | (3 << 25); // 8B, pad 4dw/128dw
        g1[1] = (int)((n8 & 0xFFFFu) << 16);                   // tensor_dim0[15:0]
        g1[2] = (int)((n8 >> 16) & 0xFFFFu) | (1 << 16);       // dim0 hi | tensor_dim1=1
        g1[3] = (int)((n8 & 0xFFFFu) << 16);                   // tile_dim0 = n8
        g1[4] = 0;                                             // tile_dim1/2 unused
        g1[5] = (int)n8;                                       // tensor_dim0_stride
        g1[6] = 0; g1[7] = 0;
        i32x4 z4; z4[0] = 0; z4[1] = 0; z4[2] = 0; z4[3] = 0;
        i32x8 z8;
#pragma unroll
        for (int i = 0; i < 8; ++i) z8[i] = 0;
        // 6-arg form (clang-23 / therock-10.0 headers): extra int32x8 group
        __builtin_amdgcn_tensor_load_to_lds(g0, g1, z4, z4, z8, 0);
        __builtin_amdgcn_s_wait_tensorcnt(0);
    }
#else
    // Fallback: manual staged copy with identical padded layout.
    for (int i = threadIdx.x; i < (HDIM * DMODEL) / 8; i += (int)blockDim.x) {
        int byteoff = (i << 4) + ((i >> 5) << 4);              // 16*i + 16*(i/32)
        *(v8bf*)((char*)wlds + byteoff) = *(const v8bf*)&wt[(size_t)i * 8];
    }
#endif
    __syncthreads();
}

// ---------------------------------------------------------------------------
// Conversion / transpose kernels
// ---------------------------------------------------------------------------
__global__ void __launch_bounds__(256) cvt_x_kernel(const float* __restrict__ x,
                                                    bf16_t* __restrict__ xb, int n) {
    int i = (blockIdx.x * 256 + threadIdx.x) * 4;
    if (i + 3 < n) {
        float4 v = *(const float4*)(x + i);
        v4bf o;
        o[0] = (bf16_t)v.x; o[1] = (bf16_t)v.y; o[2] = (bf16_t)v.z; o[3] = (bf16_t)v.w;
        *(v4bf*)(xb + i) = o;
    }
}

// Wq/Wk/Wv [H][D][HD] f32  ->  WT [3][H][HD][D] bf16 (transposed per head)
__global__ void __launch_bounds__(256) cvt_wqkv_kernel(const float* __restrict__ Wq,
                                                       const float* __restrict__ Wk,
                                                       const float* __restrict__ Wv,
                                                       bf16_t* __restrict__ WT) {
    int i = blockIdx.x * 256 + threadIdx.x;          // over H*D*HD = 589824
    int z = blockIdx.y;
    const float* W = (z == 0) ? Wq : (z == 1) ? Wk : Wv;
    int hd = i % HDIM;
    int d  = (i / HDIM) % DMODEL;
    int h  = i / (HDIM * DMODEL);
    WT[(((size_t)z * NHEAD + h) * HDIM + hd) * DMODEL + d] = (bf16_t)W[i];
}

// Wo [D][D] f32 -> WoT [D][D] bf16 transposed
__global__ void __launch_bounds__(256) cvt_wo_kernel(const float* __restrict__ Wo,
                                                     bf16_t* __restrict__ WoT) {
    int i = blockIdx.x * 256 + threadIdx.x;          // over 768*768
    int nn = i % DMODEL;
    int kk = i / DMODEL;
    WoT[(size_t)nn * DMODEL + kk] = (bf16_t)Wo[i];
}

// ---------------------------------------------------------------------------
// QKV projection: Q/K row-major [H][N][64] bf16, V transposed [H][64][N] bf16
// grid (32, 12, 3); block 256 = 8 waves; each wave: 16 rows x 64 cols.
// Whole 96KB weight tile TDM-staged into padded LDS once; barrier-free K loop.
// ---------------------------------------------------------------------------
__global__ void __launch_bounds__(256) proj_kernel(const bf16_t* __restrict__ xb,
                                                   const bf16_t* __restrict__ WT,
                                                   const float* __restrict__ bq,
                                                   const float* __restrict__ bk,
                                                   const float* __restrict__ bv,
                                                   bf16_t* __restrict__ Q,
                                                   bf16_t* __restrict__ K,
                                                   bf16_t* __restrict__ Vt) {
    __shared__ __align__(16) char wlds_raw[WTILE_BYTES];
    bf16_t* wlds = (bf16_t*)wlds_raw;

    const int lane   = threadIdx.x & 31;
    const int wave   = threadIdx.x >> 5;
    const int lanelo = lane & 15;
    const int laneHi = lane >> 4;
    const int h = blockIdx.y, z = blockIdx.z;
    const int qbase = (blockIdx.x * 8 + wave) * 16;

    const bf16_t* wt = WT + (((size_t)z * NHEAD + h) * HDIM) * DMODEL;  // [64][768]
    const float* bias = ((z == 0) ? bq : (z == 1) ? bk : bv) + h * HDIM;

    stage_wtile(wlds, wt);

    v8f acc[4];
#pragma unroll
    for (int t = 0; t < 4; ++t) acc[t] = (v8f)0.0f;

    for (int kc = 0; kc < DMODEL; kc += 32) {
        v16bf a = load_frag(&xb[(size_t)(qbase + lanelo) * DMODEL + kc + 8 * laneHi]);
#pragma unroll
        for (int t = 0; t < 4; ++t) {
            int e = (t * 16 + lanelo) * DMODEL + kc + 8 * laneHi;
            v16bf b = load_frag2(lds_pad_ptr(wlds, e), lds_pad_ptr(wlds, e + 16));
            acc[t] = wmma_bf16(a, b, acc[t]);
        }
    }

#pragma unroll
    for (int t = 0; t < 4; ++t) {
        float bcol = bias[t * 16 + lanelo];
        if (z == 2) {
            // Vt[h][hd][seq]: lane-fixed hd column, 8 consecutive seq -> 1 store
            v8bf pk;
#pragma unroll
            for (int r = 0; r < 8; ++r) pk[r] = (bf16_t)(acc[t][r] + bcol);
            *(v8bf*)&Vt[((size_t)h * HDIM + t * 16 + lanelo) * NSEQ + qbase + 8 * laneHi] = pk;
        } else {
            bf16_t* dst = (z == 0) ? Q : K;
#pragma unroll
            for (int r = 0; r < 8; ++r)
                dst[((size_t)h * NSEQ + qbase + 8 * laneHi + r) * HDIM + t * 16 + lanelo] =
                    (bf16_t)(acc[t][r] + bcol);
        }
    }
}

// ---------------------------------------------------------------------------
// Flash attention: grid (64, 12); block 128 = 4 waves, one 16-row q-tile each.
// Per 32-KV chunk: 4 WMMA (QK^T) + 4 WMMA (PV), online softmax between.
// ---------------------------------------------------------------------------
__global__ void __launch_bounds__(128) attn_kernel(const bf16_t* __restrict__ Q,
                                                   const bf16_t* __restrict__ K,
                                                   const bf16_t* __restrict__ Vt,
                                                   bf16_t* __restrict__ O) {
    __shared__ bf16_t plds[4][16 * 32];              // per-wave P staging, 4 KB

    const int lane   = threadIdx.x & 31;
    const int wave   = threadIdx.x >> 5;
    const int lanelo = lane & 15;
    const int laneHi = lane >> 4;
    const int h = blockIdx.y;
    const bool causal = (h < NLOOKB);
    const int qbase = (blockIdx.x * 4 + wave) * 16;

    const bf16_t* qh  = Q  + (size_t)h * NSEQ * HDIM;
    const bf16_t* kh  = K  + (size_t)h * NSEQ * HDIM;
    const bf16_t* vth = Vt + (size_t)h * HDIM * NSEQ;
    bf16_t* pbuf = &plds[wave][0];

    v16bf qa[2];
#pragma unroll
    for (int d = 0; d < 2; ++d)
        qa[d] = load_frag(&qh[(size_t)(qbase + lanelo) * HDIM + d * 32 + 8 * laneHi]);

    v8f oacc[4];
#pragma unroll
    for (int t = 0; t < 4; ++t) oacc[t] = (v8f)0.0f;
    float mrow[8], lrow[8];
#pragma unroll
    for (int r = 0; r < 8; ++r) { mrow[r] = -3.0e38f; lrow[r] = 0.0f; }

    const int kstart = causal ? 0 : (qbase & ~31);
    const int kend   = causal ? (qbase + 16) : NSEQ;

    for (int kbase = kstart; kbase < kend; kbase += 32) {
        // prefetch next KV tile while this one computes (global_prefetch_b8)
        if (kbase + 32 < kend) {
            __builtin_prefetch(&kh[(size_t)(kbase + 32 + lanelo) * HDIM], 0, 1);
            __builtin_prefetch(&vth[(size_t)lanelo * NSEQ + kbase + 32], 0, 1);
        }

        v8f s0 = (v8f)0.0f, s1 = (v8f)0.0f;
#pragma unroll
        for (int d = 0; d < 2; ++d) {
            v16bf b0 = load_frag(&kh[(size_t)(kbase + lanelo) * HDIM + d * 32 + 8 * laneHi]);
            v16bf b1 = load_frag(&kh[(size_t)(kbase + 16 + lanelo) * HDIM + d * 32 + 8 * laneHi]);
            s0 = wmma_bf16(qa[d], b0, s0);
            s1 = wmma_bf16(qa[d], b1, s1);
        }

        const float scl = 0.125f;                    // 1/sqrt(64)
        const int n0 = kbase + lanelo, n1 = n0 + 16;
        float tmax[8];
#pragma unroll
        for (int r = 0; r < 8; ++r) {
            int qi = qbase + 8 * laneHi + r;
            float v0 = s0[r] * scl, v1 = s1[r] * scl;
            bool ok0 = causal ? (n0 <= qi) : (n0 >= qi);
            bool ok1 = causal ? (n1 <= qi) : (n1 >= qi);
            v0 = ok0 ? v0 : -3.0e38f;
            v1 = ok1 ? v1 : -3.0e38f;
            s0[r] = v0; s1[r] = v1;
            tmax[r] = fmaxf(v0, v1);
        }
#pragma unroll
        for (int r = 0; r < 8; ++r) {
#pragma unroll
            for (int off = 1; off < 16; off <<= 1)
                tmax[r] = fmaxf(tmax[r], __shfl_xor(tmax[r], off, 32));
        }
#pragma unroll
        for (int r = 0; r < 8; ++r) {
            float mnew  = fmaxf(mrow[r], tmax[r]);
            float alpha = __expf(mrow[r] - mnew);
            float p0 = __expf(s0[r] - mnew);
            float p1 = __expf(s1[r] - mnew);
            float ps = p0 + p1;
#pragma unroll
            for (int off = 1; off < 16; off <<= 1) ps += __shfl_xor(ps, off, 32);
            lrow[r] = lrow[r] * alpha + ps;
            mrow[r] = mnew;
            s0[r] = p0; s1[r] = p1;
#pragma unroll
            for (int t = 0; t < 4; ++t) oacc[t][r] *= alpha;
        }

        // C-fragment layout -> A-fragment layout via per-wave LDS tile
#pragma unroll
        for (int r = 0; r < 8; ++r) {
            int m = 8 * laneHi + r;
            pbuf[m * 32 + lanelo]      = (bf16_t)s0[r];
            pbuf[m * 32 + 16 + lanelo] = (bf16_t)s1[r];
        }
        v16bf pa = load_frag(&pbuf[lanelo * 32 + 8 * laneHi]);
#pragma unroll
        for (int t = 0; t < 4; ++t) {
            v16bf b = load_frag(&vth[(size_t)(t * 16 + lanelo) * NSEQ + kbase + 8 * laneHi]);
            oacc[t] = wmma_bf16(pa, b, oacc[t]);
        }
    }

#pragma unroll
    for (int r = 0; r < 8; ++r) {
        float inv = 1.0f / lrow[r];
#pragma unroll
        for (int t = 0; t < 4; ++t) oacc[t][r] *= inv;
    }
    // write O concat-heads [N][768]
#pragma unroll
    for (int t = 0; t < 4; ++t)
#pragma unroll
        for (int r = 0; r < 8; ++r)
            O[(size_t)(qbase + 8 * laneHi + r) * DMODEL + h * HDIM + t * 16 + lanelo] =
                (bf16_t)oacc[t][r];
}

// ---------------------------------------------------------------------------
// Output projection: out[N][768] f32 = O[N][768]bf16 * Wo + bo
// grid (32, 12); block 256 = 8 waves; wave: 16 rows x 64-col slab of N.
// 96KB WoT slab TDM-staged once; barrier-free K loop.
// ---------------------------------------------------------------------------
__global__ void __launch_bounds__(256) oproj_kernel(const bf16_t* __restrict__ O,
                                                    const bf16_t* __restrict__ WoT,
                                                    const float* __restrict__ bo,
                                                    float* __restrict__ out) {
    __shared__ __align__(16) char wlds_raw[WTILE_BYTES];
    bf16_t* wlds = (bf16_t*)wlds_raw;

    const int lane   = threadIdx.x & 31;
    const int wave   = threadIdx.x >> 5;
    const int lanelo = lane & 15;
    const int laneHi = lane >> 4;
    const int qbase = (blockIdx.x * 8 + wave) * 16;
    const int nbase = blockIdx.y * 64;

    stage_wtile(wlds, &WoT[(size_t)nbase * DMODEL]);   // rows nbase..nbase+63

    v8f acc[4];
#pragma unroll
    for (int t = 0; t < 4; ++t) acc[t] = (v8f)0.0f;

    for (int kc = 0; kc < DMODEL; kc += 32) {
        v16bf a = load_frag(&O[(size_t)(qbase + lanelo) * DMODEL + kc + 8 * laneHi]);
#pragma unroll
        for (int t = 0; t < 4; ++t) {
            int e = (t * 16 + lanelo) * DMODEL + kc + 8 * laneHi;
            v16bf b = load_frag2(lds_pad_ptr(wlds, e), lds_pad_ptr(wlds, e + 16));
            acc[t] = wmma_bf16(a, b, acc[t]);
        }
    }

#pragma unroll
    for (int t = 0; t < 4; ++t) {
        float bcol = bo[nbase + t * 16 + lanelo];
#pragma unroll
        for (int r = 0; r < 8; ++r)
            out[(size_t)(qbase + 8 * laneHi + r) * DMODEL + nbase + t * 16 + lanelo] =
                acc[t][r] + bcol;
    }
}

// ---------------------------------------------------------------------------
extern "C" void kernel_launch(void* const* d_in, const int* in_sizes, int n_in,
                              void* d_out, int out_size, void* d_ws, size_t ws_size,
                              hipStream_t stream) {
    const float* x  = (const float*)d_in[0];
    const float* Wq = (const float*)d_in[1];
    const float* bq = (const float*)d_in[2];
    const float* Wk = (const float*)d_in[3];
    const float* bk = (const float*)d_in[4];
    const float* Wv = (const float*)d_in[5];
    const float* bv = (const float*)d_in[6];
    const float* Wo = (const float*)d_in[7];
    const float* bo = (const float*)d_in[8];
    float* out = (float*)d_out;

    char* ws = (char*)d_ws;
    bf16_t* xb  = (bf16_t*)(ws + 0);                 //  6,291,456  x bf16
    bf16_t* WT  = (bf16_t*)(ws + 6291456);           //  3,538,944  WqkvT bf16
    bf16_t* WoT = (bf16_t*)(ws + 9830400);           //  1,179,648  WoT bf16
    bf16_t* Qb  = (bf16_t*)(ws + 11010048);          //  6,291,456
    bf16_t* Kb  = (bf16_t*)(ws + 17301504);          //  6,291,456
    bf16_t* Vt  = (bf16_t*)(ws + 23592960);          //  6,291,456
    bf16_t* Ob  = (bf16_t*)(ws + 29884416);          //  6,291,456  (end 36,175,872)

    cvt_x_kernel   <<<dim3(3072),      dim3(256), 0, stream>>>(x, xb, NSEQ * DMODEL);
    cvt_wqkv_kernel<<<dim3(2304, 3),   dim3(256), 0, stream>>>(Wq, Wk, Wv, WT);
    cvt_wo_kernel  <<<dim3(2304),      dim3(256), 0, stream>>>(Wo, WoT);
    proj_kernel    <<<dim3(32, 12, 3), dim3(256), 0, stream>>>(xb, WT, bq, bk, bv, Qb, Kb, Vt);
    attn_kernel    <<<dim3(64, 12),    dim3(128), 0, stream>>>(Qb, Kb, Vt, Ob);
    oproj_kernel   <<<dim3(32, 12),    dim3(256), 0, stream>>>(Ob, WoT, bo, out);
}
